// CrossAttention_65369402245246
// MI455X (gfx1250) — matrix operand order repaired and checked by reading, not compiled, not run on hardware
//
#include <hip/hip_runtime.h>
#include <hip/hip_bf16.h>
#include <stdint.h>

// ---------------------------------------------------------------------------
// CDNA5 (gfx1250) cross-attention: all GEMMs + attention on v_wmma_f32_16x16x32_bf16
// Round 2: software-pipelined (double-buffered) GEMM core, 32x64 per-wave tiles.
// ---------------------------------------------------------------------------

typedef __attribute__((ext_vector_type(16))) __bf16 v16bf;
typedef __attribute__((ext_vector_type(8)))  __bf16 v8bf;
typedef __attribute__((ext_vector_type(8)))  float  v8f;

#define DIM_  1024
#define H_    16
#define D_    64
#define B_    2
#define N_    4096
#define M_    1024
#define EPS_  1e-6f

__device__ __forceinline__ v16bf frag_combine(v8bf lo, v8bf hi) {
  v16bf r;
#pragma unroll
  for (int t = 0; t < 8; ++t) { r[t] = lo[t]; r[8 + t] = hi[t]; }
  return r;
}

__device__ __forceinline__ v8f wmma_bf16(v16bf a, v16bf b, v8f c) {
  return __builtin_amdgcn_wmma_f32_16x16x32_bf16(false, a, false, b, (short)0, c,
                                                 false, false);
}

// Load A frags (2 row-strips) + B frags (4 col-blocks) for one K-step of 32.
__device__ __forceinline__ void load_ab(const __bf16* __restrict__ ar0,
                                        const __bf16* __restrict__ ar1,
                                        const __bf16* __restrict__ bcol,
                                        int k0, int lhalf, v16bf A[2], v16bf B[4]) {
  const int ao = k0 + 16 * lhalf;
  A[0] = frag_combine(*(const v8bf*)(ar0 + ao), *(const v8bf*)(ar0 + ao + 8));
  A[1] = frag_combine(*(const v8bf*)(ar1 + ao), *(const v8bf*)(ar1 + ao + 8));
  const int bo = k0 + 16 * lhalf;
#pragma unroll
  for (int j = 0; j < 4; ++j)
    B[j] = *(const v16bf*)(bcol + (size_t)j * 16 * 1024 + bo);
}

__device__ __forceinline__ void wmma16(const v16bf A[2], const v16bf B[4],
                                       v8f acc[2][4]) {
#pragma unroll
  for (int j = 0; j < 4; ++j) {
    acc[0][j] = wmma_bf16(A[0], B[j], acc[0][j]);
    acc[1][j] = wmma_bf16(A[1], B[j], acc[1][j]);
  }
}

// 128x64 WG tile GEMM: each wave owns 32 rows x 64 cols, K=1024, double-buffered.
__device__ __forceinline__ void gemm_core_1024(const __bf16* __restrict__ ar0,
                                               const __bf16* __restrict__ ar1,
                                               const __bf16* __restrict__ bcol,
                                               v8f acc[2][4], int lhalf) {
  v16bf Ac[2], Bc[4], An[2], Bn[4];
  load_ab(ar0, ar1, bcol, 0, lhalf, Ac, Bc);
  int k0 = 0;
  for (; k0 < 960; k0 += 64) {
    load_ab(ar0, ar1, bcol, k0 + 32, lhalf, An, Bn);   // in flight during...
    wmma16(Ac, Bc, acc);                               // ...these 8 WMMAs
    load_ab(ar0, ar1, bcol, k0 + 64, lhalf, Ac, Bc);
    wmma16(An, Bn, acc);
  }
  // k0 == 960: one K-step pair left (960 done above loads Ac for 1024? no: loop
  // exits with Ac loaded for k0=960)
  load_ab(ar0, ar1, bcol, 992, lhalf, An, Bn);
  wmma16(Ac, Bc, acc);
  wmma16(An, Bn, acc);
}

// Butterfly sum / max across the 16-lane halves (rows live per-half in C layout)
__device__ __forceinline__ float half_sum(float v) {
#pragma unroll
  for (int m = 1; m <= 8; m <<= 1) v += __shfl_xor(v, m, 32);
  return v;
}
__device__ __forceinline__ float half_max(float v) {
#pragma unroll
  for (int m = 1; m <= 8; m <<= 1) v = fmaxf(v, __shfl_xor(v, m, 32));
  return v;
}

// ---------------------------------------------------------------------------
__global__ void cvt_f32_to_bf16(const float* __restrict__ in, __bf16* __restrict__ out,
                                int n) {
  int i = blockIdx.x * 256 + threadIdx.x;
  if (i < n) out[i] = (__bf16)in[i];
}

// ---------------------------------------------------------------------------
// Q = rmsnorm_head(x @ Wq^T + bq) -> bf16 (B,H,N,D)
__global__ __launch_bounds__(128) void qproj_kernel(
    const __bf16* __restrict__ X,     // (B*N, 1024)
    const __bf16* __restrict__ W,     // (1024, 1024)
    const float* __restrict__ bias,   // (1024)
    const float* __restrict__ gamma,  // (H,D)=(1024)
    __bf16* __restrict__ Q) {
  const int wave = threadIdx.x >> 5;
  const int lane = threadIdx.x & 31;
  const int lmod = lane & 15, lhalf = lane >> 4;
  const int row0 = blockIdx.x * 128 + wave * 32;  // global (b*N + n)
  const int h = blockIdx.y;                       // head == 64-col tile
  const int col0 = h * 64;

  v8f acc[2][4] = {};
  const __bf16* ar0 = X + (size_t)(row0 + lmod) * 1024;
  const __bf16* ar1 = X + (size_t)(row0 + 16 + lmod) * 1024;
  const __bf16* bcol = W + (size_t)(col0 + lmod) * 1024;
  gemm_core_1024(ar0, ar1, bcol, acc, lhalf);

  float bj[4], gj[4];
#pragma unroll
  for (int j = 0; j < 4; ++j) {
    bj[j] = bias[col0 + 16 * j + lmod];
    gj[j] = gamma[col0 + 16 * j + lmod];
  }
#pragma unroll
  for (int s = 0; s < 2; ++s) {
#pragma unroll
    for (int j = 0; j < 4; ++j)
#pragma unroll
      for (int i = 0; i < 8; ++i) acc[s][j][i] += bj[j];
    float r[8];
#pragma unroll
    for (int i = 0; i < 8; ++i) {
      float ss = acc[s][0][i] * acc[s][0][i] + acc[s][1][i] * acc[s][1][i] +
                 acc[s][2][i] * acc[s][2][i] + acc[s][3][i] * acc[s][3][i];
      ss = half_sum(ss);
      r[i] = __frsqrt_rn(ss * (1.0f / 64.0f) + EPS_);
    }
#pragma unroll
    for (int j = 0; j < 4; ++j)
#pragma unroll
      for (int i = 0; i < 8; ++i) {
        int rowg = row0 + 16 * s + i + 8 * lhalf;
        int b = rowg >> 12, n = rowg & (N_ - 1);
        float v = acc[s][j][i] * r[i] * gj[j];
        Q[(((size_t)b * H_ + h) * N_ + n) * D_ + 16 * j + lmod] = (__bf16)v;
      }
  }
}

// ---------------------------------------------------------------------------
// KV = context @ Wkv^T + bkv ; cols [0,1024) -> K (rmsnorm, (B,H,M,D)),
//                               cols [1024,2048) -> V transposed (B,H,D,M)
__global__ __launch_bounds__(128) void kvproj_kernel(
    const __bf16* __restrict__ C,     // (B*M, 1024)
    const __bf16* __restrict__ W,     // (2048, 1024)
    const float* __restrict__ bias,   // (2048)
    const float* __restrict__ gk,     // (H,D)
    __bf16* __restrict__ Kb,          // (B,H,M,D)
    __bf16* __restrict__ Vt) {        // (B,H,D,M)
  const int wave = threadIdx.x >> 5;
  const int lane = threadIdx.x & 31;
  const int lmod = lane & 15, lhalf = lane >> 4;
  const int row0 = blockIdx.x * 128 + wave * 32;  // global (b*M + m)
  const int ct = blockIdx.y;                      // 0..31
  const int col0 = ct * 64;

  v8f acc[2][4] = {};
  const __bf16* ar0 = C + (size_t)(row0 + lmod) * 1024;
  const __bf16* ar1 = C + (size_t)(row0 + 16 + lmod) * 1024;
  const __bf16* bcol = W + (size_t)(col0 + lmod) * 1024;
  gemm_core_1024(ar0, ar1, bcol, acc, lhalf);

#pragma unroll
  for (int j = 0; j < 4; ++j) {
    float bj = bias[col0 + 16 * j + lmod];
#pragma unroll
    for (int s = 0; s < 2; ++s)
#pragma unroll
      for (int i = 0; i < 8; ++i) acc[s][j][i] += bj;
  }

  if (ct < H_) {  // K path: rmsnorm with gk
    const int h = ct;
    float gj[4];
#pragma unroll
    for (int j = 0; j < 4; ++j) gj[j] = gk[h * D_ + 16 * j + lmod];
#pragma unroll
    for (int s = 0; s < 2; ++s) {
      float r[8];
#pragma unroll
      for (int i = 0; i < 8; ++i) {
        float ss = acc[s][0][i] * acc[s][0][i] + acc[s][1][i] * acc[s][1][i] +
                   acc[s][2][i] * acc[s][2][i] + acc[s][3][i] * acc[s][3][i];
        ss = half_sum(ss);
        r[i] = __frsqrt_rn(ss * (1.0f / 64.0f) + EPS_);
      }
#pragma unroll
      for (int j = 0; j < 4; ++j)
#pragma unroll
        for (int i = 0; i < 8; ++i) {
          int rowg = row0 + 16 * s + i + 8 * lhalf;
          int b = rowg >> 10, m = rowg & (M_ - 1);
          float v = acc[s][j][i] * r[i] * gj[j];
          Kb[(((size_t)b * H_ + h) * M_ + m) * D_ + 16 * j + lmod] = (__bf16)v;
        }
    }
  } else {  // V path: transposed store
    const int h = ct - H_;
#pragma unroll
    for (int s = 0; s < 2; ++s)
#pragma unroll
      for (int j = 0; j < 4; ++j)
#pragma unroll
        for (int i = 0; i < 8; ++i) {
          int rowg = row0 + 16 * s + i + 8 * lhalf;
          int b = rowg >> 10, m = rowg & (M_ - 1);
          int d = 16 * j + lmod;
          Vt[(((size_t)b * H_ + h) * D_ + d) * M_ + m] = (__bf16)acc[s][j][i];
        }
  }
}

// ---------------------------------------------------------------------------
// Flash attention: one (b,h) x 64 query rows per block; single pass over M=1024.
__global__ __launch_bounds__(128) void attn_kernel(
    const __bf16* __restrict__ Q,    // (B,H,N,D)
    const __bf16* __restrict__ Kb,   // (B,H,M,D)
    const __bf16* __restrict__ Vt,   // (B,H,D,M)
    const unsigned char* __restrict__ mask,  // (B,M) bool
    __bf16* __restrict__ O) {        // (B,N,DIM)
  __shared__ float sbias[M_];
  __shared__ __align__(32) __bf16 sP[4][16 * 32];

  const int wave = threadIdx.x >> 5;
  const int lane = threadIdx.x & 31;
  const int lmod = lane & 15, lhalf = lane >> 4;
  const int bh = blockIdx.x;            // b*H + h
  const int b = bh >> 4, h = bh & 15;
  const int n0 = blockIdx.y * 64 + wave * 16;

  for (int m = threadIdx.x; m < M_; m += 128)
    sbias[m] = mask[b * M_ + m] ? 0.0f : -1.0e30f;
  __syncthreads();

  // Q fragments (16 rows x 64 d) : two A-frags of K=32
  const __bf16* qrow = Q + (((size_t)bh) * N_ + n0 + lmod) * D_;
  v16bf qa[2];
#pragma unroll
  for (int kk = 0; kk < 2; ++kk) {
    v8bf c0 = *(const v8bf*)(qrow + 32 * kk + 16 * lhalf);
    v8bf c1 = *(const v8bf*)(qrow + 32 * kk + 8 + 16 * lhalf);
    qa[kk] = frag_combine(c0, c1);
  }

  const __bf16* kbase = Kb + (size_t)bh * M_ * D_;
  const __bf16* vbase = Vt + (size_t)bh * D_ * M_;
  __bf16* pw = &sP[wave][0];

  v8f o[4] = {};
  float mrow[8], lrow[8];
#pragma unroll
  for (int i = 0; i < 8; ++i) { mrow[i] = -3.0e38f; lrow[i] = 0.0f; }

  for (int m0 = 0; m0 < M_; m0 += 32) {
    // S = Q * K^T for 32 keys (two 16-col tiles)
    v8f S0 = {}, S1 = {};
#pragma unroll
    for (int kk = 0; kk < 2; ++kk) {
      v16bf k0 = *(const v16bf*)(kbase + (size_t)(m0 + lmod) * D_ + 32 * kk + 16 * lhalf);
      S0 = wmma_bf16(qa[kk], k0, S0);
      v16bf k1 = *(const v16bf*)(kbase + (size_t)(m0 + 16 + lmod) * D_ + 32 * kk + 16 * lhalf);
      S1 = wmma_bf16(qa[kk], k1, S1);
    }
    // issue V fragment loads now; softmax VALU below hides their latency
    v16bf vb[4];
#pragma unroll
    for (int nb = 0; nb < 4; ++nb)
      vb[nb] = *(const v16bf*)(vbase + (size_t)(nb * 16 + lmod) * M_ + m0 + 16 * lhalf);

    const float sc = 0.125f;  // 1/sqrt(64)
    float bias0 = sbias[m0 + lmod], bias1 = sbias[m0 + 16 + lmod];
#pragma unroll
    for (int i = 0; i < 8; ++i) {
      S0[i] = S0[i] * sc + bias0;
      S1[i] = S1[i] * sc + bias1;
    }
    // online softmax
    float corr[8];
#pragma unroll
    for (int i = 0; i < 8; ++i) {
      float t = half_max(fmaxf(S0[i], S1[i]));
      float mn = fmaxf(mrow[i], t);
      corr[i] = __expf(mrow[i] - mn);
      mrow[i] = mn;
      S0[i] = __expf(S0[i] - mn);
      S1[i] = __expf(S1[i] - mn);
      float rs = half_sum(S0[i] + S1[i]);
      lrow[i] = lrow[i] * corr[i] + rs;
    }
#pragma unroll
    for (int j = 0; j < 4; ++j)
#pragma unroll
      for (int i = 0; i < 8; ++i) o[j][i] *= corr[i];

    // stage P (16x32) through wave-private LDS to reshape C-layout -> A-layout
#pragma unroll
    for (int i = 0; i < 8; ++i) {
      pw[(i + 8 * lhalf) * 32 + lmod] = (__bf16)S0[i];
      pw[(i + 8 * lhalf) * 32 + 16 + lmod] = (__bf16)S1[i];
    }
    // same-wave DS ops are in-order (DScnt), cross-wave regions are disjoint
    v8bf p0 = *(const v8bf*)(pw + lmod * 32 + 16 * lhalf);
    v8bf p1 = *(const v8bf*)(pw + lmod * 32 + 8 + 16 * lhalf);
    v16bf pa = frag_combine(p0, p1);

    // O += P * V  (V transposed -> contiguous B-frags, already loaded)
#pragma unroll
    for (int nb = 0; nb < 4; ++nb) o[nb] = wmma_bf16(pa, vb[nb], o[nb]);
  }

  float inv[8];
#pragma unroll
  for (int i = 0; i < 8; ++i) inv[i] = 1.0f / fmaxf(lrow[i], 1e-30f);
#pragma unroll
  for (int nb = 0; nb < 4; ++nb)
#pragma unroll
    for (int i = 0; i < 8; ++i) {
      int n = n0 + i + 8 * lhalf;
      float v = o[nb][i] * inv[i];
      O[((size_t)b * N_ + n) * DIM_ + h * D_ + nb * 16 + lmod] = (__bf16)v;
    }
}

// ---------------------------------------------------------------------------
// out = attn_out @ Wo^T + bo (fp32 output)
__global__ __launch_bounds__(128) void oproj_kernel(
    const __bf16* __restrict__ A,    // (B*N, 1024)
    const __bf16* __restrict__ W,    // (1024, 1024)
    const float* __restrict__ bias,  // (1024)
    float* __restrict__ Out) {       // (B*N, 1024)
  const int wave = threadIdx.x >> 5;
  const int lane = threadIdx.x & 31;
  const int lmod = lane & 15, lhalf = lane >> 4;
  const int row0 = blockIdx.x * 128 + wave * 32;
  const int col0 = blockIdx.y * 64;

  v8f acc[2][4] = {};
  const __bf16* ar0 = A + (size_t)(row0 + lmod) * 1024;
  const __bf16* ar1 = A + (size_t)(row0 + 16 + lmod) * 1024;
  const __bf16* bcol = W + (size_t)(col0 + lmod) * 1024;
  gemm_core_1024(ar0, ar1, bcol, acc, lhalf);

#pragma unroll
  for (int j = 0; j < 4; ++j) {
    float bj = bias[col0 + 16 * j + lmod];
#pragma unroll
    for (int s = 0; s < 2; ++s)
#pragma unroll
      for (int i = 0; i < 8; ++i) {
        int rowg = row0 + 16 * s + i + 8 * lhalf;
        Out[(size_t)rowg * 1024 + col0 + 16 * j + lmod] = acc[s][j][i] + bj;
      }
  }
}

// ---------------------------------------------------------------------------
extern "C" void kernel_launch(void* const* d_in, const int* in_sizes, int n_in,
                              void* d_out, int out_size, void* d_ws, size_t ws_size,
                              hipStream_t stream) {
  const float* x    = (const float*)d_in[0];   // (B,N,1024)
  const float* ctx  = (const float*)d_in[1];   // (B,M,1024)
  const unsigned char* cmask = (const unsigned char*)d_in[2];  // (B,M) bool
  const float* Wq   = (const float*)d_in[3];
  const float* bq   = (const float*)d_in[4];
  const float* Wkv  = (const float*)d_in[5];
  const float* bkv  = (const float*)d_in[6];
  const float* gq   = (const float*)d_in[7];
  const float* gk   = (const float*)d_in[8];
  const float* Wo   = (const float*)d_in[9];
  const float* bo   = (const float*)d_in[10];
  float* out = (float*)d_out;

  char* ws = (char*)d_ws;
  __bf16* xb   = (__bf16*)(ws + 0);          // 16 MB  (8192x1024)
  __bf16* cb   = (__bf16*)(ws + 16777216);   //  4 MB  (2048x1024)
  __bf16* wqb  = (__bf16*)(ws + 20971520);   //  2 MB
  __bf16* wkvb = (__bf16*)(ws + 23068672);   //  4 MB
  __bf16* wob  = (__bf16*)(ws + 27262976);   //  2 MB
  __bf16* qb   = (__bf16*)(ws + 29360128);   // 16 MB  (B,H,N,D)
  __bf16* kb   = (__bf16*)(ws + 46137344);   //  4 MB  (B,H,M,D)
  __bf16* vtb  = (__bf16*)(ws + 50331648);   //  4 MB  (B,H,D,M)
  __bf16* aob  = (__bf16*)(ws + 54525952);   // 16 MB  (B,N,1024)

  const int T = 256;
  cvt_f32_to_bf16<<<(B_ * N_ * DIM_ + T - 1) / T, T, 0, stream>>>(x, xb, B_ * N_ * DIM_);
  cvt_f32_to_bf16<<<(B_ * M_ * DIM_ + T - 1) / T, T, 0, stream>>>(ctx, cb, B_ * M_ * DIM_);
  cvt_f32_to_bf16<<<(DIM_ * DIM_ + T - 1) / T, T, 0, stream>>>(Wq, wqb, DIM_ * DIM_);
  cvt_f32_to_bf16<<<(2 * DIM_ * DIM_ + T - 1) / T, T, 0, stream>>>(Wkv, wkvb, 2 * DIM_ * DIM_);
  cvt_f32_to_bf16<<<(DIM_ * DIM_ + T - 1) / T, T, 0, stream>>>(Wo, wob, DIM_ * DIM_);

  qproj_kernel<<<dim3(B_ * N_ / 128, H_), 128, 0, stream>>>(xb, wqb, bq, gq, qb);
  kvproj_kernel<<<dim3(B_ * M_ / 128, 2 * H_), 128, 0, stream>>>(cb, wkvb, bkv, gk, kb, vtb);
  attn_kernel<<<dim3(B_ * H_, N_ / 64), 128, 0, stream>>>(qb, kb, vtb, cmask, aob);
  oproj_kernel<<<dim3(B_ * N_ / 128, DIM_ / 64), 128, 0, stream>>>(aob, wob, bo, out);
}
